// ContextAttention_38388417692113
// MI455X (gfx1250) — matrix operand order repaired and checked by reading, not compiled, not run on hardware
//
#include <hip/hip_runtime.h>
#include <hip/hip_bf16.h>

#define T 9216
#define HH 96
#define WW 96

typedef __attribute__((ext_vector_type(16))) __bf16 v16bf;
typedef __attribute__((ext_vector_type(8)))  __bf16 v8bf;
typedef __attribute__((ext_vector_type(8)))  float  v8f;
typedef __attribute__((ext_vector_type(4)))  int    v4i;

#if defined(__AMDGCN__) && __has_builtin(__builtin_amdgcn_global_load_async_to_lds_b128)
#define HAVE_ASYNC_LDS 1
#else
#define HAVE_ASYNC_LDS 0
#endif

#if HAVE_ASYNC_LDS
typedef __attribute__((address_space(1))) v4i as1_v4i;
typedef __attribute__((address_space(3))) v4i as3_v4i;
__device__ __forceinline__ void async_copy16(const void* g, void* l) {
    __builtin_amdgcn_global_load_async_to_lds_b128((as1_v4i*)g, (as3_v4i*)l, 0, 0);
}
__device__ __forceinline__ void wait_asynccnt0() {
#if __has_builtin(__builtin_amdgcn_s_wait_asynccnt)
    __builtin_amdgcn_s_wait_asynccnt(0);
#else
    asm volatile("s_wait_asynccnt 0" ::: "memory");
#endif
}
#endif

// ---------------- conv1: 1 -> 64, 3x3, pad 1, relu ----------------
__global__ __launch_bounds__(256) void conv1_relu(const float* __restrict__ x,
    const float* __restrict__ w1, const float* __restrict__ b1,
    float* __restrict__ f1)
{
    int idx = blockIdx.x * 256 + threadIdx.x;
    int p   = idx % T;
    int c   = (idx / T) & 63;
    int img = idx / (T * 64);
    int y = p / WW, xp = p % WW;
    float acc = b1[c];
    const float* in = x + (size_t)img * T;
    const float* w  = w1 + c * 9;
    #pragma unroll
    for (int ky = 0; ky < 3; ++ky) {
        int iy = y + ky - 1;
        if ((unsigned)iy >= (unsigned)HH) continue;
        #pragma unroll
        for (int kx = 0; kx < 3; ++kx) {
            int ix = xp + kx - 1;
            if ((unsigned)ix >= (unsigned)WW) continue;
            acc += w[ky * 3 + kx] * in[iy * WW + ix];
        }
    }
    f1[(size_t)(img * 64 + c) * T + p] = fmaxf(acc, 0.f);
}

// ---------------- conv2: 64 -> 64, 3x3, pad 1, relu; out token-major [img][t][64]
__global__ __launch_bounds__(256) void conv2_relu(const float* __restrict__ f1,
    const float* __restrict__ w2, const float* __restrict__ b2,
    float* __restrict__ f2)
{
    int idx  = blockIdx.x * 256 + threadIdx.x;
    int p    = idx % T;
    int rest = idx / T;
    int cog  = rest & 7;
    int img  = rest >> 3;
    int co0  = cog * 8;
    int y = p / WW, xp = p % WW;
    float acc[8];
    #pragma unroll
    for (int j = 0; j < 8; ++j) acc[j] = b2[co0 + j];
    for (int ci = 0; ci < 64; ++ci) {
        const float* base = f1 + (size_t)(img * 64 + ci) * T;
        #pragma unroll
        for (int ky = 0; ky < 3; ++ky) {
            int iy = y + ky - 1;
            if ((unsigned)iy >= (unsigned)HH) continue;
            #pragma unroll
            for (int kx = 0; kx < 3; ++kx) {
                int ix = xp + kx - 1;
                if ((unsigned)ix >= (unsigned)WW) continue;
                float v = base[iy * WW + ix];
                #pragma unroll
                for (int j = 0; j < 8; ++j)
                    acc[j] += w2[((co0 + j) * 64 + ci) * 9 + ky * 3 + kx] * v;
            }
        }
    }
    float* out = f2 + ((size_t)img * T + p) * 64 + co0;
    #pragma unroll
    for (int j = 0; j < 8; ++j) out[j] = fmaxf(acc[j], 0.f);
}

// ---------------- QKV projection: feat channel = c*2+n; write bf16 [b][t][64]
__global__ __launch_bounds__(256) void qkv_proj(const float* __restrict__ f2,
    const float* __restrict__ qw, const float* __restrict__ qb,
    const float* __restrict__ kw, const float* __restrict__ kb,
    const float* __restrict__ vw, const float* __restrict__ vb,
    __bf16* __restrict__ Q, __bf16* __restrict__ K, __bf16* __restrict__ V)
{
    int idx = blockIdx.x * 256 + threadIdx.x;
    int o = idx & 63;
    int t = (idx >> 6) % T;
    int b = idx / (T * 64);
    float aq = qb[o], ak = kb[o], av = vb[o];
    #pragma unroll
    for (int n = 0; n < 2; ++n) {
        const float* fp = f2 + ((size_t)(b * 2 + n) * T + t) * 64;
        for (int c = 0; c < 64; ++c) {
            float f = fp[c];
            int ic = c * 2 + n;
            aq += qw[o * 128 + ic] * f;
            ak += kw[o * 128 + ic] * f;
            av += vw[o * 128 + ic] * f;
        }
    }
    size_t off = ((size_t)b * T + t) * 64 + o;
    Q[off] = (__bf16)aq; K[off] = (__bf16)ak; V[off] = (__bf16)av;
}

// ---------------- flash attention: 4 waves/block, 64 queries/block ----------------
// Double-buffered K/V tiles; K staged via async global->LDS when available.
__global__ __launch_bounds__(128) void flash_attn(const __bf16* __restrict__ Qg,
    const __bf16* __restrict__ Kg, const __bf16* __restrict__ Vg,
    float* __restrict__ out)
{
    __shared__ __attribute__((aligned(16))) __bf16 Kt[2][64 * 64];   // [key][c]
    __shared__ __attribute__((aligned(16))) __bf16 Vt[2][64 * 64];   // [c][key] (transposed)
    __shared__ __attribute__((aligned(16))) __bf16 Pbuf[4][16 * 64]; // per-wave P tile [m][n]

    const int tid  = threadIdx.x;
    const int wave = tid >> 5;
    const int lane = tid & 31;
    const int col  = lane & 15;
    const int half = lane >> 4;

    const int b     = blockIdx.x / 144;
    const int qbase = (blockIdx.x % 144) * 64;

    // Q A-fragments for this wave: rows m = col; A16x32: k = (e>>3)*16 + half*8 + (e&7)
    const __bf16* qp = Qg + ((size_t)b * T + qbase + wave * 16 + col) * 64;
    v16bf aq0, aq1;
    {
        v8bf l0 = *(const v8bf*)(qp + half * 8);
        v8bf l1 = *(const v8bf*)(qp + 16 + half * 8);
        v8bf l2 = *(const v8bf*)(qp + 32 + half * 8);
        v8bf l3 = *(const v8bf*)(qp + 48 + half * 8);
        #pragma unroll
        for (int i = 0; i < 8; ++i) {
            aq0[i] = l0[i]; aq0[8 + i] = l1[i];
            aq1[i] = l2[i]; aq1[8 + i] = l3[i];
        }
    }

    float mrow[8], lrow[8];
    v8f Oacc[4];
    #pragma unroll
    for (int r = 0; r < 8; ++r) { mrow[r] = -1e30f; lrow[r] = 0.f; }
    #pragma unroll
    for (int cj = 0; cj < 4; ++cj) Oacc[cj] = (v8f)(0.0f);

    const __bf16* Kbase = Kg + (size_t)b * T * 64;
    const __bf16* Vbase = Vg + (size_t)b * T * 64;
    const float scale = 0.125f; // 1/sqrt(64)

    const int srow  = tid >> 1;        // 0..63 : staging row (key index in tile)
    const int scoff = (tid & 1) * 32;  // 0 or 32 : channel offset

    // ---- prologue: stage tile 0 into buffer 0 ----
    {
        const __bf16* ks = Kbase + (size_t)srow * 64 + scoff;
        const __bf16* vs = Vbase + (size_t)srow * 64 + scoff;
#if HAVE_ASYNC_LDS
        #pragma unroll
        for (int i = 0; i < 4; ++i)
            async_copy16(ks + i * 8, &Kt[0][srow * 64 + scoff + i * 8]);
#else
        #pragma unroll
        for (int i = 0; i < 4; ++i)
            *(v8bf*)&Kt[0][srow * 64 + scoff + i * 8] = *(const v8bf*)(ks + i * 8);
#endif
        #pragma unroll
        for (int i = 0; i < 4; ++i) {
            v8bf vv = *(const v8bf*)(vs + i * 8);
            #pragma unroll
            for (int e = 0; e < 8; ++e)
                Vt[0][(scoff + i * 8 + e) * 64 + srow] = vv[e];
        }
#if HAVE_ASYNC_LDS
        wait_asynccnt0();
#endif
    }
    __syncthreads();

    for (int kv = 0; kv < 144; ++kv) {
        const int cur = kv & 1, nxt = cur ^ 1;
        const bool have_next = (kv + 1) < 144;

        // ---- issue staging for next tile (overlaps with compute below) ----
        v8bf vreg[4];
#if !HAVE_ASYNC_LDS
        v8bf kreg[4];
#endif
        if (have_next) {
            const __bf16* ks = Kbase + (size_t)((kv + 1) * 64 + srow) * 64 + scoff;
            const __bf16* vs = Vbase + (size_t)((kv + 1) * 64 + srow) * 64 + scoff;
#if HAVE_ASYNC_LDS
            #pragma unroll
            for (int i = 0; i < 4; ++i)
                async_copy16(ks + i * 8, &Kt[nxt][srow * 64 + scoff + i * 8]);
#else
            #pragma unroll
            for (int i = 0; i < 4; ++i)
                kreg[i] = *(const v8bf*)(ks + i * 8);
#endif
            #pragma unroll
            for (int i = 0; i < 4; ++i)
                vreg[i] = *(const v8bf*)(vs + i * 8);
            if (kv + 2 < 144) {
                __builtin_prefetch(ks + 2 * 64 * 64, 0, 1);
                __builtin_prefetch(vs + 2 * 64 * 64, 0, 1);
            }
        }

        const __bf16* KtC = Kt[cur];
        const __bf16* VtC = Vt[cur];

        // ---- S = Q * K^T : 4 key sub-tiles, 2 wmma each (K-dim 64 = 2x32) ----
        v8f S[4];
        #pragma unroll
        for (int j = 0; j < 4; ++j) {
            const __bf16* kp = KtC + (j * 16 + col) * 64; // B: n = key = col, k = channel
            v16bf bk0, bk1;
            {
                v8bf c0 = *(const v8bf*)(kp + half * 8);
                v8bf c1 = *(const v8bf*)(kp + 16 + half * 8);
                v8bf c2 = *(const v8bf*)(kp + 32 + half * 8);
                v8bf c3 = *(const v8bf*)(kp + 48 + half * 8);
                #pragma unroll
                for (int i = 0; i < 8; ++i) {
                    bk0[i] = c0[i]; bk0[8 + i] = c1[i];
                    bk1[i] = c2[i]; bk1[8 + i] = c3[i];
                }
            }
            v8f acc = {};
            acc = __builtin_amdgcn_wmma_f32_16x16x32_bf16(false, aq0, false, bk0, (short)0, acc, false, false);
            acc = __builtin_amdgcn_wmma_f32_16x16x32_bf16(false, aq1, false, bk1, (short)0, acc, false, false);
            S[j] = acc;
        }

        // ---- online softmax (rows live in half-wave lane groups) ----
        #pragma unroll
        for (int r = 0; r < 8; ++r) {
            float s0 = S[0][r] * scale, s1 = S[1][r] * scale,
                  s2 = S[2][r] * scale, s3 = S[3][r] * scale;
            float mx = fmaxf(fmaxf(s0, s1), fmaxf(s2, s3));
            #pragma unroll
            for (int off = 1; off < 16; off <<= 1)
                mx = fmaxf(mx, __shfl_xor(mx, off, 16));
            float mnew  = fmaxf(mrow[r], mx);
            float alpha = __expf(mrow[r] - mnew);
            mrow[r] = mnew;
            float p0 = __expf(s0 - mnew), p1 = __expf(s1 - mnew);
            float p2 = __expf(s2 - mnew), p3 = __expf(s3 - mnew);
            int m = r + half * 8;
            __bf16* pb = &Pbuf[wave][m * 64 + col];
            pb[0]  = (__bf16)p0; pb[16] = (__bf16)p1;
            pb[32] = (__bf16)p2; pb[48] = (__bf16)p3;
            float ls = p0 + p1 + p2 + p3;
            #pragma unroll
            for (int off = 1; off < 16; off <<= 1)
                ls += __shfl_xor(ls, off, 16);
            lrow[r] = lrow[r] * alpha + ls;
            #pragma unroll
            for (int cj = 0; cj < 4; ++cj) Oacc[cj][r] *= alpha;
        }
        asm volatile("" ::: "memory"); // keep LDS store->load order (same wave, in-order DS)

        // ---- reload P in A-layout, then O += P * V ----
        const __bf16* pp = &Pbuf[wave][col * 64];
        v16bf ap0, ap1;
        {
            v8bf l0 = *(const v8bf*)(pp + half * 8);
            v8bf l1 = *(const v8bf*)(pp + 16 + half * 8);
            v8bf l2 = *(const v8bf*)(pp + 32 + half * 8);
            v8bf l3 = *(const v8bf*)(pp + 48 + half * 8);
            #pragma unroll
            for (int i = 0; i < 8; ++i) {
                ap0[i] = l0[i]; ap0[8 + i] = l1[i];
                ap1[i] = l2[i]; ap1[8 + i] = l3[i];
            }
        }
        #pragma unroll
        for (int cj = 0; cj < 4; ++cj) {
            const __bf16* vp = VtC + (cj * 16 + col) * 64; // B: n = channel = col, k = key
            v16bf bv0, bv1;
            {
                v8bf c0 = *(const v8bf*)(vp + half * 8);
                v8bf c1 = *(const v8bf*)(vp + 16 + half * 8);
                v8bf c2 = *(const v8bf*)(vp + 32 + half * 8);
                v8bf c3 = *(const v8bf*)(vp + 48 + half * 8);
                #pragma unroll
                for (int i = 0; i < 8; ++i) {
                    bv0[i] = c0[i]; bv0[8 + i] = c1[i];
                    bv1[i] = c2[i]; bv1[8 + i] = c3[i];
                }
            }
            Oacc[cj] = __builtin_amdgcn_wmma_f32_16x16x32_bf16(false, ap0, false, bv0, (short)0, Oacc[cj], false, false);
            Oacc[cj] = __builtin_amdgcn_wmma_f32_16x16x32_bf16(false, ap1, false, bv1, (short)0, Oacc[cj], false, false);
        }

        // ---- commit next-tile staging, then publish ----
        if (have_next) {
#if !HAVE_ASYNC_LDS
            #pragma unroll
            for (int i = 0; i < 4; ++i)
                *(v8bf*)&Kt[nxt][srow * 64 + scoff + i * 8] = kreg[i];
#endif
            #pragma unroll
            for (int i = 0; i < 4; ++i) {
                #pragma unroll
                for (int e = 0; e < 8; ++e)
                    Vt[nxt][(scoff + i * 8 + e) * 64 + srow] = vreg[i][e];
            }
#if HAVE_ASYNC_LDS
            wait_asynccnt0();
#endif
        }
        __syncthreads();
    }

    // ---- epilogue: normalize rows, write ctx [b][ch][t] ----
    #pragma unroll
    for (int r = 0; r < 8; ++r) {
        float inv = 1.f / lrow[r];
        int tok = qbase + wave * 16 + r + half * 8;
        #pragma unroll
        for (int cj = 0; cj < 4; ++cj) {
            int ch = cj * 16 + col;
            out[((size_t)b * 64 + ch) * T + tok] = Oacc[cj][r] * inv;
        }
    }
}

extern "C" void kernel_launch(void* const* d_in, const int* in_sizes, int n_in,
                              void* d_out, int out_size, void* d_ws, size_t ws_size,
                              hipStream_t stream)
{
    const float* frames = (const float*)d_in[0];
    const float* w1 = (const float*)d_in[1];
    const float* b1 = (const float*)d_in[2];
    const float* w2 = (const float*)d_in[3];
    const float* b2 = (const float*)d_in[4];
    const float* qw = (const float*)d_in[5];
    const float* qb = (const float*)d_in[6];
    const float* kw = (const float*)d_in[7];
    const float* kb = (const float*)d_in[8];
    const float* vw = (const float*)d_in[9];
    const float* vb = (const float*)d_in[10];

    char* ws = (char*)d_ws;
    float*  f1 = (float*)ws;                                   // 4*64*9216*4  = 9,437,184 B
    float*  f2 = (float*)(ws + 9437184);                       // 9,437,184 B
    __bf16* Qb = (__bf16*)(ws + 2 * 9437184);                  // 2,359,296 B
    __bf16* Kb = (__bf16*)(ws + 2 * 9437184 + 2359296);        // 2,359,296 B
    __bf16* Vb = (__bf16*)(ws + 2 * 9437184 + 2 * 2359296);    // 2,359,296 B

    conv1_relu<<<(4 * 64 * T) / 256, 256, 0, stream>>>(frames, w1, b1, f1);
    conv2_relu<<<(4 * 8 * T) / 256, 256, 0, stream>>>(f1, w2, b2, f2);
    qkv_proj<<<(2 * 64 * T) / 256, 256, 0, stream>>>(f2, qw, qb, kw, kb, vw, vb, Qb, Kb, Vb);
    flash_attn<<<2 * 144, 128, 0, stream>>>(Qb, Kb, Vb, (float*)d_out);
}
